// ChebyKANLayer_81303730913619
// MI455X (gfx1250) — matrix operand order repaired and checked by reading, not compiled, not run on hardware
//
#include <hip/hip_runtime.h>
#include <hip/hip_bf16.h>
#include <cstdint>

// ---------------------------------------------------------------------------
// ChebyKAN layer: y[b,o] = sum_{d,i} T_d(tanh(x[b,i]*s+t)) * W[d,i,o]
//   x: (8192,1024) f32, W: (9,1024,1024) f32, out: (8192,1024) f32
// Strategy:
//   - degree 0 (T0==1) folded into per-column bias (kernel bias_build)
//   - W[1..8] transposed+converted to bf16 Wt[d][o][i] in workspace
//     (B-operand of WMMA wants per-lane consecutive K=i)
//   - fused GEMM: tanh + Chebyshev recurrence in f32 regs -> bf16 A tiles in
//     LDS (ISA A-layout, 80B padded rows) -> v_wmma_f32_16x16x32_bf16
// ---------------------------------------------------------------------------

#define I_DIM 1024
#define O_DIM 1024
#define BATCH 8192
#define NDEG  8            // degrees 1..8 through WMMA
#define KC    32           // K chunk (bf16 WMMA K)
#define ROWB  80           // basis LDS row stride in bytes (40 bf16, bank-safe)
#define DSTRIDE (64 * ROWB)

typedef __attribute__((ext_vector_type(16))) __bf16 v16bf;
typedef __attribute__((ext_vector_type(8)))  float  v8f;

struct Frag32 { uint4 lo, hi; };

__device__ __forceinline__ v16bf as_frag(uint4 lo, uint4 hi) {
  Frag32 f{lo, hi};
  return __builtin_bit_cast(v16bf, f);
}

__device__ __forceinline__ uint32_t bf16rne(float f) {
  uint32_t u = __builtin_bit_cast(uint32_t, f);
  return (u + 0x7FFFu + ((u >> 16) & 1u)) >> 16;
}

__device__ __forceinline__ uint32_t pack_bf16(float a, float b) {
  return bf16rne(a) | (bf16rne(b) << 16);
}

// ---- kernel 1: bias[o] = sum_i W[0,i,o]  (degree-0 contribution) ----------
__global__ void bias_build(const float* __restrict__ W, float* __restrict__ bias) {
  const int o = blockIdx.x * blockDim.x + threadIdx.x;
  float s = 0.f;
  for (int i = 0; i < I_DIM; ++i) s += W[(size_t)i * O_DIM + o];
  bias[o] = s;
}

// ---- kernel 2: Wt[d][o][i] = bf16(W[d+1][i][o]), d = 0..7 ------------------
__global__ void wt_build(const float* __restrict__ W, uint16_t* __restrict__ wt) {
  __shared__ float tile[32][33];
  const int d  = blockIdx.z;          // maps to degree d+1
  const int i0 = blockIdx.x * 32;
  const int o0 = blockIdx.y * 32;
#pragma unroll
  for (int r = 0; r < 32; r += 8)
    tile[threadIdx.y + r][threadIdx.x] =
        W[((size_t)(d + 1) * I_DIM + (i0 + threadIdx.y + r)) * O_DIM + (o0 + threadIdx.x)];
  __syncthreads();
#pragma unroll
  for (int r = 0; r < 32; r += 8) {
    float v = tile[threadIdx.x][threadIdx.y + r];
    wt[((size_t)d * O_DIM + (o0 + threadIdx.y + r)) * I_DIM + (i0 + threadIdx.x)] =
        (uint16_t)bf16rne(v);
  }
}

// ---- kernel 3: fused tanh + Chebyshev + bf16 WMMA GEMM ---------------------
// grid (O/128, BATCH/64), 256 threads = 8 waves; wave tile 32x32 (2x2 WMMA)
__global__ __launch_bounds__(256, 2)
void cheby_gemm(const float* __restrict__ x,
                const uint16_t* __restrict__ wt,     // [8][O][I] bf16
                const float* __restrict__ bias0,     // [O]
                const float* __restrict__ tanh_scale,
                const float* __restrict__ tanh_bias,
                float* __restrict__ out) {
  __shared__ __align__(16) unsigned char basis[NDEG * DSTRIDE];  // 40 KB

  const int tid  = threadIdx.x;
  const int lane = tid & 31;
  const int w    = tid >> 5;
  const int wm   = w >> 2;                 // 0..1  (M wave row)
  const int wn   = w & 3;                  // 0..3  (N wave col)
  const int m0   = blockIdx.y * 64;
  const int n0   = blockIdx.x * 128;
  const int mw   = wm * 32;                // local M base of wave
  const int nw   = n0 + wn * 32;           // global N base of wave

  const float ts = tanh_scale[0];
  const float tb = tanh_bias[0];

  // accumulators, initialized with degree-0 bias (same column for all 8 rows)
  v8f acc[2][2];
#pragma unroll
  for (int nt = 0; nt < 2; ++nt) {
    float bv = bias0[nw + nt * 16 + (lane & 15)];
    v8f a;
#pragma unroll
    for (int j = 0; j < 8; ++j) a[j] = bv;
    acc[0][nt] = a;
    acc[1][nt] = a;
  }

  // staging map: thread -> (row sr, k-quarter sq) of the 64x32 x-chunk
  const int sr = tid >> 2;                 // 0..63
  const int sq = tid & 3;                  // k offset = sq*8
  const float* xrow = x + (size_t)(m0 + sr) * I_DIM + sq * 8;
  unsigned char* bdst = basis + sr * ROWB + sq * 16;

  // per-lane A-fragment base: lane 0-15 row m, K 0-7/16-23; lane 16-31 K 8-15/24-31
  const unsigned char* abase =
      basis + (mw + (lane & 15)) * ROWB + ((lane >> 4) << 4);

  for (int kc = 0; kc < I_DIM / KC; ++kc) {
    const int kbase = kc * KC;

    // ---- tanh + Chebyshev recurrence for this chunk (registers) ----
    float4 xa = *(const float4*)(xrow + kbase);
    float4 xb = *(const float4*)(xrow + kbase + 4);
    float xt[8], tp[8], tc[8];
    xt[0] = xa.x; xt[1] = xa.y; xt[2] = xa.z; xt[3] = xa.w;
    xt[4] = xb.x; xt[5] = xb.y; xt[6] = xb.z; xt[7] = xb.w;
#pragma unroll
    for (int j = 0; j < 8; ++j) {
      xt[j] = tanhf(xt[j] * ts + tb);
      tp[j] = 1.0f;          // T0
      tc[j] = xt[j];         // T1
    }
    if (kc + 1 < I_DIM / KC) __builtin_prefetch(xrow + kbase + KC, 0, 1);

    __syncthreads();  // previous chunk fully consumed

#pragma unroll
    for (int d = 0; d < NDEG; ++d) {       // store T_{d+1}, advance recurrence
      uint4 p;
      p.x = pack_bf16(tc[0], tc[1]);
      p.y = pack_bf16(tc[2], tc[3]);
      p.z = pack_bf16(tc[4], tc[5]);
      p.w = pack_bf16(tc[6], tc[7]);
      *(uint4*)(bdst + d * DSTRIDE) = p;
#pragma unroll
      for (int j = 0; j < 8; ++j) {
        float tn = 2.0f * xt[j] * tc[j] - tp[j];
        tp[j] = tc[j];
        tc[j] = tn;
      }
    }

    __syncthreads();  // basis tiles visible

    // ---- WMMA over 8 degrees, accumulate into the same C ----
#pragma unroll
    for (int d = 0; d < NDEG; ++d) {
      const unsigned char* ab = abase + d * DSTRIDE;
      v16bf a0 = as_frag(*(const uint4*)(ab),
                         *(const uint4*)(ab + 32));
      v16bf a1 = as_frag(*(const uint4*)(ab + 16 * ROWB),
                         *(const uint4*)(ab + 16 * ROWB + 32));
#pragma unroll
      for (int nt = 0; nt < 2; ++nt) {
        const int ncol = nw + nt * 16 + (lane & 15);
        const unsigned char* bb = (const unsigned char*)wt +
            (((size_t)d * O_DIM + ncol) * I_DIM + kbase + ((lane >> 4) << 3)) * 2;
        v16bf bf = as_frag(*(const uint4*)(bb), *(const uint4*)(bb + 32));
        acc[0][nt] = __builtin_amdgcn_wmma_f32_16x16x32_bf16(
            false, a0, false, bf, (short)0, acc[0][nt], false, false);
        acc[1][nt] = __builtin_amdgcn_wmma_f32_16x16x32_bf16(
            false, a1, false, bf, (short)0, acc[1][nt], false, false);
      }
    }
  }

  // ---- write C: VGPR r -> row r (lanes 0-15) / r+8 (lanes 16-31) ----
#pragma unroll
  for (int mt = 0; mt < 2; ++mt) {
    const int mb = m0 + mw + mt * 16 + ((lane >> 4) << 3);
#pragma unroll
    for (int nt = 0; nt < 2; ++nt) {
      const int n = nw + nt * 16 + (lane & 15);
#pragma unroll
      for (int r = 0; r < 8; ++r)
        out[(size_t)(mb + r) * O_DIM + n] = acc[mt][nt][r];
    }
  }
}

// ---------------------------------------------------------------------------
extern "C" void kernel_launch(void* const* d_in, const int* in_sizes, int n_in,
                              void* d_out, int out_size, void* d_ws, size_t ws_size,
                              hipStream_t stream) {
  const float* x      = (const float*)d_in[0];
  const float* W      = (const float*)d_in[1];   // (9,1024,1024) f32
  const float* tscale = (const float*)d_in[2];
  const float* tbias  = (const float*)d_in[3];
  float* out = (float*)d_out;

  // workspace: [0,4KB) bias; [8KB, 8KB+16MB) Wt bf16 (8 x 1024 x 1024)
  float*    bias = (float*)d_ws;
  uint16_t* wtq  = (uint16_t*)((char*)d_ws + 8192);

  bias_build<<<O_DIM / 256, 256, 0, stream>>>(W, bias);
  wt_build<<<dim3(I_DIM / 32, O_DIM / 32, NDEG), dim3(32, 8), 0, stream>>>(W, wtq);
  cheby_gemm<<<dim3(O_DIM / 128, BATCH / 64), 256, 0, stream>>>(
      x, wtq, bias, tscale, tbias, out);
}
